// QuantumAutoencoder_52578989638124
// MI455X (gfx1250) — compile-verified, matrix operand order checked by reading, not run on hardware
//
#include <hip/hip_runtime.h>
#include <hip/hip_bf16.h>

typedef __attribute__((ext_vector_type(16))) _Float16 v16h;
typedef __attribute__((ext_vector_type(8)))  float    v8f;
typedef __attribute__((ext_vector_type(4)))  unsigned u32x4;
typedef __attribute__((ext_vector_type(4)))  int      i32x4;
typedef __attribute__((ext_vector_type(8)))  int      i32x8;

#define NQ 12
#define DIM 4096
#define NL 2
#define B_ROWS 4096
#define D_IN 1024
#define H_DIM 48

__device__ __forceinline__ float2 cmul(float2 a, float2 b) {
    return make_float2(a.x * b.x - a.y * b.y, a.x * b.y + a.y * b.x);
}
__device__ __forceinline__ float2 cadd(float2 a, float2 b) {
    return make_float2(a.x + b.x, a.y + b.y);
}

// K index within a 32-deep fragment for (laneGroup g, vgpr v) pair base.
__device__ __forceinline__ int frag_k(int g, int v) {
    return (v < 4) ? (g * 8 + 2 * v) : (16 + g * 8 + 2 * (v - 4));
}

// ---------------------------------------------------------------------------
// TDM 1-D async copy: n8 elements of 8 bytes, global -> LDS.
// D# per CDNA5 ISA ch.8: group0 {count=1, lds_addr, global_addr[56:0], type=2},
// group1 {data_size=3(8B), tensor_dim0=n8, tensor_dim1=1, tile_dim0=n8,
//         stride0=n8}, groups 2/3 zero (<=2D tensor, VADDR2/3 still supplied).
// Tracked by TENSORcnt; waited with s_wait_tensorcnt.
// ---------------------------------------------------------------------------
__device__ __forceinline__ void tdm_load_1d(unsigned lds_addr, const void* gsrc,
                                            unsigned n8) {
    unsigned long long ga = (unsigned long long)(uintptr_t)gsrc;
    u32x4 g0 = { 1u,                                   // count=1 (valid user D#)
                 lds_addr,                             // LDS byte address
                 (unsigned)ga,                         // global_addr[31:0]
                 ((unsigned)(ga >> 32) & 0x01FFFFFFu)  // global_addr[56:32]
                 | 0x80000000u };                      // type=2 ("image")
    i32x8 g1 = { (int)(3u << 16),                      // data_size=8B
                 (int)((n8 & 0xFFFFu) << 16),          // tensor_dim0[15:0]
                 (int)(((n8 >> 16) & 0xFFFFu) | (1u << 16)), // tdim0 hi | tdim1=1
                 (int)((n8 & 0xFFFFu) << 16),          // tile_dim0 = n8
                 0,                                    // tile_dim1/2 = 0
                 (int)n8,                              // tensor_dim0_stride lo
                 0, 0 };
    i32x4 g2 = {0, 0, 0, 0};
    i32x4 g3 = {0, 0, 0, 0};
    asm volatile("tensor_load_to_lds %0, %1, %2, %3"
                 :: "s"(g0), "s"(g1), "s"(g2), "s"(g3)
                 : "memory");
}

__device__ __forceinline__ unsigned lds_off(const void* p) {
    return (unsigned)(uintptr_t)p;   // low 32 bits of flat LDS addr = LDS offset
}

// ---------------------------------------------------------------------------
// Pack an f32 row-major [K x N] weight matrix into per-lane B-fragment layout:
// dst[ ((ktile*NT + ntile)*32 + lane)*16 + h ]  (16 halves contiguous/lane).
// ---------------------------------------------------------------------------
__global__ void qae_pack_b(const float* __restrict__ src, _Float16* __restrict__ dst,
                           int NT, int N, int total) {
    int idx = blockIdx.x * blockDim.x + threadIdx.x;
    if (idx >= total) return;
    int h = idx & 15;
    int lane = (idx >> 4) & 31;
    int tile = idx >> 9;
    int ntile = tile % NT;
    int ktile = tile / NT;
    int g = lane >> 4;
    int v = h >> 1;
    int p = h & 1;
    int k = ktile * 32 + frag_k(g, v) + p;
    int n = ntile * 16 + (lane & 15);
    dst[idx] = (_Float16)src[k * N + n];
}

// ---------------------------------------------------------------------------
// Encoder: lat[B,12] = (relu(x@W1+b1)) @ W2 + b2.  One wave per 16-row tile.
// Packed W1 streamed through LDS by the Tensor Data Mover, double-buffered:
// chunk = 8 K-steps x 3 n-tiles = 24 KB, 4 chunks total.
// ---------------------------------------------------------------------------
__global__ void __launch_bounds__(32)
qae_encoder(const float* __restrict__ x, const float* __restrict__ b1,
            const float* __restrict__ w2, const float* __restrict__ b2,
            const _Float16* __restrict__ pw1, float* __restrict__ lat) {
    __shared__ __align__(32) _Float16 bstage[2][8 * 3 * 512];  // 2 x 24 KB
    __shared__ float hs[16][H_DIM];
    const int lane = threadIdx.x;
    const int g = lane >> 4;
    const int mrow = lane & 15;
    const int row0 = blockIdx.x * 16;
    const float* xrow = x + (size_t)(row0 + mrow) * D_IN;

    v8f acc[3];
#pragma unroll
    for (int nt = 0; nt < 3; ++nt) acc[nt] = (v8f){0.f,0.f,0.f,0.f,0.f,0.f,0.f,0.f};

    const unsigned CHUNK_H = 8 * 3 * 512;          // halves per chunk
    tdm_load_1d(lds_off(&bstage[0][0]), pw1, CHUNK_H / 4);   // 3072 x 8B

    for (int c = 0; c < 4; ++c) {
        if (c < 3) {
            tdm_load_1d(lds_off(&bstage[(c + 1) & 1][0]),
                        pw1 + (size_t)(c + 1) * CHUNK_H, CHUNK_H / 4);
            __builtin_amdgcn_s_wait_tensorcnt(1);  // chunk c landed, c+1 in flight
        } else {
            __builtin_amdgcn_s_wait_tensorcnt(0);
        }
        asm volatile("" ::: "memory");
        const _Float16* bb = &bstage[c & 1][0];

        for (int kt8 = 0; kt8 < 8; ++kt8) {
            const int kt = c * 8 + kt8;
            v16h a;
#pragma unroll
            for (int v = 0; v < 8; ++v) {
                int kk = kt * 32 + frag_k(g, v);
                float2 xv = *(const float2*)(xrow + kk);
                a[2 * v]     = (_Float16)xv.x;
                a[2 * v + 1] = (_Float16)xv.y;
            }
#pragma unroll
            for (int nt = 0; nt < 3; ++nt) {
                v16h b = *(const v16h*)(bb + ((size_t)(kt8 * 3 + nt) * 32 + lane) * 16);
                acc[nt] = __builtin_amdgcn_wmma_f32_16x16x32_f16(
                    false, a, false, b, (short)0, acc[nt], false, false);
            }
        }
    }
    // bias + relu -> LDS tile
#pragma unroll
    for (int nt = 0; nt < 3; ++nt) {
#pragma unroll
        for (int r = 0; r < 8; ++r) {
            int m = r + 8 * g;
            int n = nt * 16 + mrow;
            hs[m][n] = fmaxf(acc[nt][r] + b1[n], 0.0f);
        }
    }
    __syncthreads();
    // tiny GEMM2: 16x48 @ 48x12 ; 6 outputs per lane
#pragma unroll
    for (int e = 0; e < 6; ++e) {
        int o = lane * 6 + e;
        int m = o / NQ;
        int c = o % NQ;
        float s = b2[c];
#pragma unroll 8
        for (int k = 0; k < H_DIM; ++k) s += hs[m][k] * w2[k * NQ + c];
        lat[(size_t)(row0 + m) * NQ + c] = s;
    }
}

// ---------------------------------------------------------------------------
// Quantum sim: one block per sample; statevector lives in LDS.
// RX*RY*RZ fused to one 2x2 complex gate per (layer,qubit); CNOT chain fused
// to a single amplitude permutation (prefix-xor of bits) per layer.
// ---------------------------------------------------------------------------
__global__ void __launch_bounds__(256)
qae_qsim(const float* __restrict__ lat, const float* __restrict__ cp,
         float* __restrict__ latq) {
    __shared__ float2 st[DIM];
    __shared__ float2 gates[36][4];   // 0..11 encode RY, 12..35 fused RZ*RY*RX
    __shared__ float wsum[8][NQ];

    const int tid = threadIdx.x;
    const int sample = blockIdx.x;

    if (tid < NQ) {                       // per-sample RY encode gates
        float ang = lat[(size_t)sample * NQ + tid];
        float c = cosf(0.5f * ang), s = sinf(0.5f * ang);
        gates[tid][0] = make_float2(c, 0.f);
        gates[tid][1] = make_float2(-s, 0.f);
        gates[tid][2] = make_float2(s, 0.f);
        gates[tid][3] = make_float2(c, 0.f);
    } else if (tid < 36) {                // fused variational gates
        int t = tid - NQ;
        int base = t * 3;
        float ca = cosf(0.5f * cp[base]),     sa = sinf(0.5f * cp[base]);
        float cb = cosf(0.5f * cp[base + 1]), sb = sinf(0.5f * cp[base + 1]);
        float cg = cosf(0.5f * cp[base + 2]), sg = sinf(0.5f * cp[base + 2]);
        float2 M00 = make_float2(cb * ca,  sb * sa);
        float2 M01 = make_float2(-sb * ca, -cb * sa);
        float2 M10 = make_float2(sb * ca,  -cb * sa);
        float2 M11 = make_float2(cb * ca,  -sb * sa);
        float2 e0 = make_float2(cg, -sg);
        float2 e1 = make_float2(cg,  sg);
        gates[tid][0] = cmul(e0, M00);
        gates[tid][1] = cmul(e0, M01);
        gates[tid][2] = cmul(e1, M10);
        gates[tid][3] = cmul(e1, M11);
    }
#pragma unroll
    for (int j = 0; j < DIM / 256; ++j) {
        int i = tid + 256 * j;
        st[i] = (i == 0) ? make_float2(1.f, 0.f) : make_float2(0.f, 0.f);
    }
    __syncthreads();

    auto apply_gate = [&](int gi, int q) {
        float2 g00 = gates[gi][0], g01 = gates[gi][1];
        float2 g10 = gates[gi][2], g11 = gates[gi][3];
#pragma unroll
        for (int j = 0; j < DIM / 512; ++j) {       // 2048 pairs / 256 threads
            int p = tid + 256 * j;
            int i0 = ((p >> q) << (q + 1)) | (p & ((1 << q) - 1));
            int i1 = i0 | (1 << q);
            float2 a0 = st[i0], a1 = st[i1];
            st[i0] = cadd(cmul(g00, a0), cmul(g01, a1));
            st[i1] = cadd(cmul(g10, a0), cmul(g11, a1));
        }
        __syncthreads();
    };

    for (int q = 0; q < NQ; ++q) apply_gate(q, q);          // amplitude encode

    for (int L = 0; L < NL; ++L) {
        for (int q = 0; q < NQ; ++q) apply_gate(NQ + L * NQ + q, q);
        float2 tmp[DIM / 256];
#pragma unroll
        for (int j = 0; j < DIM / 256; ++j) {
            unsigned u = (unsigned)(tid + 256 * j);
#pragma unroll
            for (int qq = NQ - 2; qq >= 0; --qq)
                if ((u >> qq) & 1u) u ^= (2u << qq);
            tmp[j] = st[u];
        }
        __syncthreads();
#pragma unroll
        for (int j = 0; j < DIM / 256; ++j) st[tid + 256 * j] = tmp[j];
        __syncthreads();
    }

    float acc[NQ];
#pragma unroll
    for (int q = 0; q < NQ; ++q) acc[q] = 0.f;
#pragma unroll
    for (int j = 0; j < DIM / 256; ++j) {
        int i = tid + 256 * j;
        float2 a = st[i];
        float pr = a.x * a.x + a.y * a.y;
#pragma unroll
        for (int q = 0; q < NQ; ++q)
            acc[q] += ((i >> q) & 1) ? -pr : pr;
    }
#pragma unroll
    for (int q = 0; q < NQ; ++q) {
#pragma unroll
        for (int off = 16; off >= 1; off >>= 1)
            acc[q] += __shfl_xor(acc[q], off, 32);
    }
    int wave = tid >> 5, lane = tid & 31;
    if (lane == 0) {
#pragma unroll
        for (int q = 0; q < NQ; ++q) wsum[wave][q] = acc[q];
    }
    __syncthreads();
    if (tid < NQ) {
        float s = 0.f;
#pragma unroll
        for (int w = 0; w < 8; ++w) s += wsum[w][tid];
        latq[(size_t)sample * NQ + tid] = s;
    }
}

// ---------------------------------------------------------------------------
// Decoder: h2 = relu(latq@W1+b1) fused into LDS (f16), then WMMA vs dec_w2.
// Block = 8 waves; block owns 16 rows, each wave owns 128 columns.
// ---------------------------------------------------------------------------
__global__ void __launch_bounds__(256)
qae_decoder(const float* __restrict__ latq, const float* __restrict__ w1,
            const float* __restrict__ b1, const float* __restrict__ b2,
            const _Float16* __restrict__ pw2, float* __restrict__ out) {
    __shared__ __align__(32) _Float16 h2s[16 * D_IN];   // 32 KB
    __shared__ float lqs[16 * NQ];

    const int tid = threadIdx.x;
    const int row0 = blockIdx.x * 16;

    if (tid < 16 * NQ) lqs[tid] = latq[(size_t)row0 * NQ + tid];
    __syncthreads();

#pragma unroll 4
    for (int it = 0; it < (16 * D_IN) / 256; ++it) {
        int flat = tid + 256 * it;
        int m = flat >> 10;
        int j = flat & (D_IN - 1);
        float s = b1[j];
#pragma unroll
        for (int k = 0; k < NQ; ++k) s += lqs[m * NQ + k] * w1[k * D_IN + j];
        h2s[m * D_IN + j] = (_Float16)fmaxf(s, 0.0f);
    }
    __syncthreads();

    const int wave = tid >> 5;
    const int lane = tid & 31;
    const int g = lane >> 4;
    const int mrow = lane & 15;

    v8f acc[8];
#pragma unroll
    for (int t = 0; t < 8; ++t) acc[t] = (v8f){0.f,0.f,0.f,0.f,0.f,0.f,0.f,0.f};

    for (int kt = 0; kt < D_IN / 32; ++kt) {
        if (kt + 1 < D_IN / 32) {    // cover L2 latency of next B burst
            __builtin_prefetch(pw2 + ((size_t)((kt + 1) * 64 + wave * 8) * 32) * 16, 0, 0);
        }
        v16h a;
#pragma unroll
        for (int v = 0; v < 8; ++v) {
            int kk = kt * 32 + frag_k(g, v);
            a[2 * v]     = h2s[mrow * D_IN + kk];
            a[2 * v + 1] = h2s[mrow * D_IN + kk + 1];
        }
#pragma unroll
        for (int t = 0; t < 8; ++t) {
            int nt = wave * 8 + t;
            v16h b = *(const v16h*)(pw2 + ((size_t)(kt * 64 + nt) * 32 + lane) * 16);
            acc[t] = __builtin_amdgcn_wmma_f32_16x16x32_f16(
                false, a, false, b, (short)0, acc[t], false, false);
        }
    }
#pragma unroll
    for (int t = 0; t < 8; ++t) {
        int n = (wave * 8 + t) * 16 + mrow;
#pragma unroll
        for (int r = 0; r < 8; ++r) {
            int m = r + 8 * g;
            out[(size_t)(row0 + m) * D_IN + n] = acc[t][r] + b2[n];
        }
    }
}

// ---------------------------------------------------------------------------
extern "C" void kernel_launch(void* const* d_in, const int* in_sizes, int n_in,
                              void* d_out, int out_size, void* d_ws, size_t ws_size,
                              hipStream_t stream) {
    const float* x      = (const float*)d_in[0];
    const float* enc_w1 = (const float*)d_in[1];
    const float* enc_b1 = (const float*)d_in[2];
    const float* enc_w2 = (const float*)d_in[3];
    const float* enc_b2 = (const float*)d_in[4];
    const float* circ   = (const float*)d_in[5];
    const float* dec_w1 = (const float*)d_in[6];
    const float* dec_b1 = (const float*)d_in[7];
    const float* dec_w2 = (const float*)d_in[8];
    const float* dec_b2 = (const float*)d_in[9];
    float* out = (float*)d_out;

    char* ws = (char*)d_ws;
    _Float16* pw1  = (_Float16*)(ws);                       // 32*3*512  halves =  96 KB
    _Float16* pw2  = (_Float16*)(ws + 98304);               // 32*64*512 halves =   2 MB
    float*    lat  = (float*)   (ws + 98304 + 2097152);     // 4096*12 f32 = 192 KB
    float*    latq = (float*)   (ws + 98304 + 2097152 + 196608);

    const int tot1 = 32 * 3 * 512;
    const int tot2 = 32 * 64 * 512;
    qae_pack_b<<<(tot1 + 255) / 256, 256, 0, stream>>>(enc_w1, pw1, 3, H_DIM, tot1);
    qae_pack_b<<<(tot2 + 255) / 256, 256, 0, stream>>>(dec_w2, pw2, 64, D_IN, tot2);

    qae_encoder<<<B_ROWS / 16, 32, 0, stream>>>(x, enc_b1, enc_w2, enc_b2, pw1, lat);

    qae_qsim<<<B_ROWS, 256, 0, stream>>>(lat, circ, latq);

    qae_decoder<<<B_ROWS / 16, 256, 0, stream>>>(latq, dec_w1, dec_b1, dec_b2, pw2, out);
}
